// Attention_1443109012366
// MI455X (gfx1250) — matrix operand order repaired and checked by reading, not compiled, not run on hardware
//
#include <hip/hip_runtime.h>

// ---------------------------------------------------------------------------
// Fused video attention block for MI455X (gfx1250, wave32, WMMA).
// Pipeline: [prep weights bf16^T] -> [QKV GEMM + scale + RoPE] ->
//           [per-(b,m,h) attention with bias/mask/softmax] -> [out proj].
// All GEMMs via v_wmma_f32_16x16x32_bf16 (fp32 accumulate).
// Each GEMM wave owns a 64x64 output block: every B-fragment is reused by
// 4 row strips, cutting weight-side L2 traffic 4x vs. 16-row blocks.
// ---------------------------------------------------------------------------

#define B_DIM      2
#define M_DIM      1024
#define N_DIM      64
#define DIM        512
#define HEADS      8
#define DHEAD      64
#define HIDDEN     512
#define HIDDEN3    1536

typedef __attribute__((ext_vector_type(16))) __bf16 bf16x16;
typedef __attribute__((ext_vector_type(8)))  __bf16 bf16x8;
typedef __attribute__((ext_vector_type(8)))  float  f32x8;
typedef int v4i __attribute__((vector_size(4 * sizeof(int))));

#if defined(__has_builtin)
#if __has_builtin(__builtin_amdgcn_global_load_async_to_lds_b128)
#define HAVE_ASYNC_LDS 1
#endif
#endif

__device__ __forceinline__ void wait_asynccnt0() {
#if defined(__has_builtin) && __has_builtin(__builtin_amdgcn_s_wait_asynccnt)
    __builtin_amdgcn_s_wait_asynccnt(0);
#else
    asm volatile("s_wait_asynccnt 0" ::: "memory");
#endif
}

__device__ __forceinline__ f32x8 wmma_bf16(bf16x16 a, bf16x16 b, f32x8 c) {
    return __builtin_amdgcn_wmma_f32_16x16x32_bf16(
        false, a, false, b, (short)0, c, false, false);
}

// A-fragment (16x32 bf16): lane L holds row M=L%16; K = {kb..kb+15},
// kb = 16*(L/16).  Source is row-major with leading dimension ld.
__device__ __forceinline__ bf16x16 load_a_frag(const __bf16* base, int ld,
                                               int koff, int lane) {
    const int row = lane & 15;
    const int kb  = (lane >> 4) * 16;
    const __bf16* p = base + row * ld + koff + kb;
    bf16x8 lo = *(const bf16x8*)(p);
    bf16x8 hi = *(const bf16x8*)(p + 8);
    bf16x16 r;
#pragma unroll
    for (int i = 0; i < 8; ++i) { r[i] = lo[i]; r[i + 8] = hi[i]; }
    return r;
}

// B-fragment (32x16 bf16): lane L holds column N=n0+L%16; K = 16 contiguous
// values starting at koff + 16*(L/16).  `bt` is B^T row-major (row n == B col n),
// so the 16 K-values are one contiguous 32-byte load.
__device__ __forceinline__ bf16x16 load_b_frag(const __bf16* bt, int ld,
                                               int n0, int koff, int lane) {
    const int n = n0 + (lane & 15);
    const int k = koff + (lane >> 4) * 16;
    return *(const bf16x16*)(bt + (size_t)n * ld + k);
}

// ---------------------------------------------------------------------------
// Kernel 0: convert + transpose weights to bf16 (one-shot, tiny).
// ---------------------------------------------------------------------------
__global__ void prep_weights_kernel(const float* __restrict__ wqkv,
                                    const float* __restrict__ wout,
                                    __bf16* __restrict__ wqkvT,
                                    __bf16* __restrict__ woutT) {
    const int idx = blockIdx.x * 256 + threadIdx.x;
    if (idx < DIM * HIDDEN3) {
        const int k = idx / HIDDEN3, n = idx % HIDDEN3;
        wqkvT[(size_t)n * DIM + k] = (__bf16)wqkv[idx];
    }
    if (idx < HIDDEN * DIM) {
        const int k = idx / DIM, n = idx % DIM;
        woutT[(size_t)n * HIDDEN + k] = (__bf16)wout[idx];
    }
}

// ---------------------------------------------------------------------------
// Kernel 1: QKV = x @ w_qkv, scale q, apply interleaved RoPE to q,k.
// One WG per (b,m) slice (64x512 @ 512x1536).  x staged once in LDS as bf16.
// Each wave computes 64 rows x 64 cols per phase (16 WMMA tiles); 8 waves x
// 3 phases cover the 24 column blocks.  B frags reused across 4 row strips.
// q,k stored row-major (n x d) per head; v stored transposed (d x n).
// ---------------------------------------------------------------------------
__global__ __launch_bounds__(256) void qkv_rope_kernel(
    const float* __restrict__ x, const __bf16* __restrict__ wqkvT,
    __bf16* __restrict__ qws, __bf16* __restrict__ kws,
    __bf16* __restrict__ vws) {
    __shared__ __align__(32) __bf16 xs[N_DIM * DIM];   // 64 KB
    const int bm  = blockIdx.x;
    const int tid = threadIdx.x;

    const float* xp = x + (size_t)bm * N_DIM * DIM;
    for (int i = tid; i < N_DIM * DIM; i += 256) xs[i] = (__bf16)xp[i];
    __syncthreads();

    const int wave = tid >> 5, lane = tid & 31;
    const int half = lane >> 4, nlo = lane & 15;

    for (int phase = 0; phase < 3; ++phase) {
        const int cb = phase * 8 + wave;           // 64-col block: 0..23
        f32x8 acc[16] = {};                        // [strip][tile]

        for (int ks = 0; ks < DIM / 32; ++ks) {
            bf16x16 a[4];
#pragma unroll
            for (int s = 0; s < 4; ++s)
                a[s] = load_a_frag(xs + s * 16 * DIM, DIM, ks * 32, lane);
#pragma unroll
            for (int t = 0; t < 4; ++t) {
                bf16x16 b = load_b_frag(wqkvT, DIM, cb * 64 + t * 16,
                                        ks * 32, lane);
#pragma unroll
                for (int s = 0; s < 4; ++s)
                    acc[s * 4 + t] = wmma_bf16(a[s], b, acc[s * 4 + t]);
            }
        }

#pragma unroll
        for (int t = 0; t < 4; ++t) {
            const int c = cb * 64 + t * 16 + nlo;  // 0..1535
            const int section = c >> 9;            // 0=q 1=k 2=v (uniform/tile)
            const int hd = c & 511, h = hd >> 6, dh = hd & 63;
            const size_t hs = (((size_t)bm * HEADS + h) * N_DIM) * DHEAD;
#pragma unroll
            for (int s = 0; s < 4; ++s) {
#pragma unroll
                for (int r = 0; r < 8; ++r) {
                    const int row = s * 16 + r + 8 * half;  // n position
                    float v = acc[s * 4 + t][r];
                    if (section == 0) v *= 0.125f;  // DIM_HEAD^-0.5
                    // rotary pair partner sits in the adjacent lane (d^1)
                    const float other = __shfl_xor(v, 1, 32);
                    if (section < 2) {
                        const float f   = (float)(dh >> 1);
                        const float inv = __powf(10000.0f, -f * (1.0f / 32.0f));
                        const float ang = (float)row * inv;
                        const float rot = (dh & 1) ? other : -other;
                        v = v * __cosf(ang) + rot * __sinf(ang);
                    }
                    if (section == 0)
                        qws[hs + (size_t)row * DHEAD + dh] = (__bf16)v;
                    else if (section == 1)
                        kws[hs + (size_t)row * DHEAD + dh] = (__bf16)v;
                    else
                        vws[hs + (size_t)dh * N_DIM + row] = (__bf16)v;
                }
            }
        }
    }
}

// ---------------------------------------------------------------------------
// Kernel 2: per-(b,m,h) attention. 128 threads; wave = 16-row strip.
// sim = q k^T (+bias, mask) -> softmax -> attn @ v^T -> O (bf16, n x 512).
// ---------------------------------------------------------------------------
__global__ __launch_bounds__(128) void attn_kernel(
    const __bf16* __restrict__ qws, const __bf16* __restrict__ kws,
    const __bf16* __restrict__ vws, const float* __restrict__ pos_bias,
    const int* __restrict__ focus_mask, __bf16* __restrict__ ows) {
    __shared__ __align__(32) __bf16 attn_s[4 * 16 * N_DIM];  // 8 KB
    const int slice = blockIdx.x;        // bm*HEADS + h
    const int h  = slice & (HEADS - 1);
    const int bm = slice >> 3;
    const int b  = bm >> 10;             // m = 1024
    const int tid = threadIdx.x, lane = tid & 31;
    const int strip = tid >> 5;
    const int half = lane >> 4, nlo = lane & 15;

    const __bf16* q = qws + (size_t)slice * N_DIM * DHEAD;
    const __bf16* k = kws + (size_t)slice * N_DIM * DHEAD;
    const __bf16* v = vws + (size_t)slice * N_DIM * DHEAD;

    // sim = q @ k^T  (k row-major == (k^T)^T, so k rows are B columns)
    f32x8 sim[4] = {};
#pragma unroll
    for (int ks = 0; ks < 2; ++ks) {
        bf16x16 a = load_a_frag(q + strip * 16 * DHEAD, DHEAD, ks * 32, lane);
#pragma unroll
        for (int t = 0; t < 4; ++t) {
            bf16x16 bb = load_b_frag(k, DHEAD, t * 16, ks * 32, lane);
            sim[t] = wmma_bf16(a, bb, sim[t]);
        }
    }

    const bool eye = (focus_mask[b] != 0);
    const float* pb = pos_bias + (size_t)h * N_DIM * N_DIM;
#pragma unroll
    for (int t = 0; t < 4; ++t) {
#pragma unroll
        for (int r = 0; r < 8; ++r) {
            const int row = strip * 16 + r + 8 * half;
            const int col = t * 16 + nlo;
            float s = sim[t][r] + pb[row * N_DIM + col];
            if (eye && (col != row)) s = -3.402823466e38f;
            sim[t][r] = s;
        }
    }

    // Row softmax: a row lives in one 16-lane half across the 4 tiles.
#pragma unroll
    for (int r = 0; r < 8; ++r) {
        float mx = fmaxf(fmaxf(sim[0][r], sim[1][r]),
                         fmaxf(sim[2][r], sim[3][r]));
#pragma unroll
        for (int m = 8; m >= 1; m >>= 1) mx = fmaxf(mx, __shfl_xor(mx, m, 32));
        float e[4], sum = 0.0f;
#pragma unroll
        for (int t = 0; t < 4; ++t) { e[t] = __expf(sim[t][r] - mx); sum += e[t]; }
#pragma unroll
        for (int m = 8; m >= 1; m >>= 1) sum += __shfl_xor(sum, m, 32);
        const float inv = 1.0f / sum;
#pragma unroll
        for (int t = 0; t < 4; ++t) sim[t][r] = e[t] * inv;
    }

    // Re-shape attn into A-fragment layout via LDS (bf16, 16x64 per strip).
    __bf16* as = attn_s + strip * 16 * N_DIM;
#pragma unroll
    for (int t = 0; t < 4; ++t)
#pragma unroll
        for (int r = 0; r < 8; ++r) {
            const int m = r + 8 * half;
            as[m * N_DIM + t * 16 + nlo] = (__bf16)sim[t][r];
        }
    __syncthreads();

    // out = attn @ v ; v stored transposed (d x n) so B-frags are contiguous.
    f32x8 o[4] = {};
#pragma unroll
    for (int ks = 0; ks < 2; ++ks) {
        bf16x16 a = load_a_frag(as, N_DIM, ks * 32, lane);
#pragma unroll
        for (int t = 0; t < 4; ++t) {
            bf16x16 bb = load_b_frag(v, N_DIM, t * 16, ks * 32, lane);
            o[t] = wmma_bf16(a, bb, o[t]);
        }
    }

    __bf16* op = ows + (size_t)bm * N_DIM * HIDDEN;
#pragma unroll
    for (int t = 0; t < 4; ++t)
#pragma unroll
        for (int r = 0; r < 8; ++r) {
            const int row  = strip * 16 + r + 8 * half;
            const int dcol = t * 16 + nlo;
            op[(size_t)row * HIDDEN + h * DHEAD + dcol] = (__bf16)o[t][r];
        }
}

// ---------------------------------------------------------------------------
// Kernel 3: final = O @ w_out. One WG per (b,m) slice (64x512 @ 512x512).
// O tile (already bf16) staged to LDS via async global->LDS when available.
// Each wave owns 64 rows x 64 cols (16 tiles); B frags reused 4x.
// ---------------------------------------------------------------------------
__global__ __launch_bounds__(256) void out_proj_kernel(
    const __bf16* __restrict__ ows, const __bf16* __restrict__ woutT,
    float* __restrict__ out) {
    __shared__ __align__(32) __bf16 os[N_DIM * HIDDEN];  // 64 KB
    const int bm = blockIdx.x, tid = threadIdx.x;
    const __bf16* op = ows + (size_t)bm * N_DIM * HIDDEN;

#if defined(HAVE_ASYNC_LDS)
    // Async DMA staging: 4096 x 16B chunks, each lane moves its own chunk.
    // Builtin expects (as1 int4* src, as3 int4* dst, imm offset, imm cpol).
    for (int i = tid; i < (N_DIM * HIDDEN) / 8; i += 256) {
        __builtin_amdgcn_global_load_async_to_lds_b128(
            (__attribute__((address_space(1))) v4i*)(op + (size_t)i * 8),
            (__attribute__((address_space(3))) v4i*)(os + i * 8), 0, 0);
    }
    wait_asynccnt0();
#else
    for (int i = tid; i < N_DIM * HIDDEN; i += 256) os[i] = op[i];
#endif
    __syncthreads();

    const int wave = tid >> 5, lane = tid & 31;
    const int cb = wave;                           // 64-col block: 0..7

    f32x8 acc[16] = {};                            // [strip][tile]
    for (int ks = 0; ks < HIDDEN / 32; ++ks) {
        bf16x16 a[4];
#pragma unroll
        for (int s = 0; s < 4; ++s)
            a[s] = load_a_frag(os + s * 16 * HIDDEN, HIDDEN, ks * 32, lane);
#pragma unroll
        for (int t = 0; t < 4; ++t) {
            bf16x16 b = load_b_frag(woutT, HIDDEN, cb * 64 + t * 16,
                                    ks * 32, lane);
#pragma unroll
            for (int s = 0; s < 4; ++s)
                acc[s * 4 + t] = wmma_bf16(a[s], b, acc[s * 4 + t]);
        }
    }

    const int half = lane >> 4, nlo = lane & 15;
    float* dst = out + (size_t)bm * N_DIM * DIM;
#pragma unroll
    for (int t = 0; t < 4; ++t)
#pragma unroll
        for (int s = 0; s < 4; ++s)
#pragma unroll
            for (int r = 0; r < 8; ++r) {
                const int row = s * 16 + r + 8 * half;
                const int c   = cb * 64 + t * 16 + nlo;
                dst[(size_t)row * DIM + c] = acc[s * 4 + t][r];
            }
}

// ---------------------------------------------------------------------------
extern "C" void kernel_launch(void* const* d_in, const int* in_sizes, int n_in,
                              void* d_out, int out_size, void* d_ws,
                              size_t ws_size, hipStream_t stream) {
    const float* x        = (const float*)d_in[0];
    const float* pos_bias = (const float*)d_in[1];
    const float* w_qkv    = (const float*)d_in[2];
    const float* w_out    = (const float*)d_in[3];
    const int*   focus    = (const int*)d_in[4];
    float* out = (float*)d_out;

    char* ws = (char*)d_ws;
    size_t off = 0;
    __bf16* wqkvT = (__bf16*)(ws + off); off += (size_t)HIDDEN3 * DIM * 2;
    __bf16* woutT = (__bf16*)(ws + off); off += (size_t)HIDDEN * DIM * 2;
    const size_t qkv_elems = (size_t)B_DIM * M_DIM * HEADS * N_DIM * DHEAD;
    __bf16* qws = (__bf16*)(ws + off); off += qkv_elems * 2;
    __bf16* kws = (__bf16*)(ws + off); off += qkv_elems * 2;
    __bf16* vws = (__bf16*)(ws + off); off += qkv_elems * 2;
    __bf16* ows = (__bf16*)(ws + off); off += (size_t)B_DIM * M_DIM * N_DIM * HIDDEN * 2;

    const int BM = B_DIM * M_DIM;  // 2048

    prep_weights_kernel<<<(DIM * HIDDEN3 + 255) / 256, 256, 0, stream>>>(
        w_qkv, w_out, wqkvT, woutT);

    qkv_rope_kernel<<<BM, 256, 0, stream>>>(x, wqkvT, qws, kws, vws);

    attn_kernel<<<BM * HEADS, 128, 0, stream>>>(
        qws, kws, vws, pos_bias, focus, ows);

    out_proj_kernel<<<BM, 256, 0, stream>>>(ows, woutT, out);
}
